// CausalSelfAttention_3161095929858
// MI455X (gfx1250) — matrix operand order repaired and checked by reading, not compiled, not run on hardware
//
#include <hip/hip_runtime.h>
#include <hip/hip_bf16.h>
#include <stdint.h>

// ---------------------------------------------------------------------------
// CDNA5 (gfx1250) causal self-attention, bf16 WMMA pipeline.
//   B=4, T=2048, C=2048, NH=16, NKV=8, HD=128
// Stages: f32->bf16 convert | fused QKV GEMM (async-LDS double-buffered) |
//         RoPE(+layout) | V transpose | flash attention (all WMMA) |
//         output GEMM (fp32 out)
// ---------------------------------------------------------------------------

#define BATCH 4
#define SEQ   2048
#define CDIM  2048
#define NH    16
#define NKV   8
#define HD    128
#define QKVN  4096   // 2048 (Q) + 1024 (K) + 1024 (V)

typedef __attribute__((ext_vector_type(16))) __bf16 v16bf;
typedef __attribute__((ext_vector_type(8)))  float  v8f;
typedef int v4i __attribute__((__vector_size__(16)));   // builtin param type

union Frag32 {            // one 32-byte WMMA A/B operand (16 x bf16)
    v16bf v;
    uint4 u[2];
};

#if __has_builtin(__builtin_amdgcn_global_load_async_to_lds_b128) && \
    __has_builtin(__builtin_amdgcn_s_wait_asynccnt)
#define HAVE_ASYNC 1
#else
#define HAVE_ASYNC 0
#endif

// async global->LDS 16B copy (GLOBAL_LOAD_ASYNC_TO_LDS_B128, ASYNCcnt);
// falls back to a VGPR round-trip copy if the builtin is absent.
__device__ __forceinline__ void async_copy16(const unsigned short* g,
                                             unsigned short* l) {
#if HAVE_ASYNC
    __builtin_amdgcn_global_load_async_to_lds_b128(
        (__attribute__((address_space(1))) v4i*)(uintptr_t)g,
        (__attribute__((address_space(3))) v4i*)(unsigned)(uintptr_t)l,
        0, 0);
#else
    *(uint4*)l = *(const uint4*)g;
#endif
}
__device__ __forceinline__ void wait_async0() {
#if HAVE_ASYNC
    __builtin_amdgcn_s_wait_asynccnt(0);
#endif
}

__device__ __forceinline__ float bf2f(unsigned short u) {
    return __uint_as_float(((unsigned)u) << 16);
}
__device__ __forceinline__ unsigned short f2bf(float f) {
    unsigned u = __float_as_uint(f);
    u += 0x7FFFu + ((u >> 16) & 1u);          // round-to-nearest-even
    return (unsigned short)(u >> 16);
}
__device__ __forceinline__ unsigned packbf2(float a, float b) {
    return (unsigned)f2bf(a) | ((unsigned)f2bf(b) << 16);
}

// ---------------------------------------------------------------------------
// fp32 -> bf16 conversion (grid-stride)
// ---------------------------------------------------------------------------
__global__ void k_cvt_bf16(const float* __restrict__ s,
                           unsigned short* __restrict__ d, long n) {
    long i = (long)blockIdx.x * blockDim.x + threadIdx.x;
    long st = (long)gridDim.x * blockDim.x;
    for (; i < n; i += st) d[i] = f2bf(s[i]);
}

// ---------------------------------------------------------------------------
// RoPE for Q: qkv[B,T,4096] cols 0..2047 -> Qr[B,NH,T,HD], 1/sqrt(HD) folded.
// cos/sin index d uses freq index d>>1 (repeat-2 cache); rotate-half pairing.
// ---------------------------------------------------------------------------
__global__ void k_rope_q(const unsigned short* __restrict__ qkv,
                         unsigned short* __restrict__ Qr) {
    const long n = (long)BATCH * SEQ * NH * HD;
    long i = (long)blockIdx.x * blockDim.x + threadIdx.x;
    long st = (long)gridDim.x * blockDim.x;
    for (; i < n; i += st) {
        int d  = (int)(i & 127);
        int hh = (int)((i >> 7) & 15);
        int t  = (int)((i >> 11) & 2047);
        int b  = (int)(i >> 22);
        size_t row = (size_t)(b * SEQ + t) * QKVN + hh * HD;
        float xv = bf2f(qkv[row + d]);
        int   dp = (d < 64) ? d + 64 : d - 64;
        float xo = bf2f(qkv[row + dp]);
        float fi = (float)(2 * (d >> 1));
        float ang = (float)t * __expf(-fi * (9.210340371976184f / 128.0f)); // 10000^(-2i/hd)
        float c = __cosf(ang), s = __sinf(ang);
        float r = (d < 64) ? (xv * c - xo * s) : (xv * c + xo * s);
        r *= 0.08838834764831845f;            // 1/sqrt(HD) attention scale
        Qr[((size_t)(b * NH + hh) * SEQ + t) * HD + d] = f2bf(r);
    }
}

// RoPE for K: qkv cols 2048..3071 -> Kr[B,NKV,T,HD]
__global__ void k_rope_k(const unsigned short* __restrict__ qkv,
                         unsigned short* __restrict__ Kr) {
    const long n = (long)BATCH * SEQ * NKV * HD;
    long i = (long)blockIdx.x * blockDim.x + threadIdx.x;
    long st = (long)gridDim.x * blockDim.x;
    for (; i < n; i += st) {
        int d  = (int)(i & 127);
        int kh = (int)((i >> 7) & 7);
        int t  = (int)((i >> 10) & 2047);
        int b  = (int)(i >> 21);
        size_t row = (size_t)(b * SEQ + t) * QKVN + 2048 + kh * HD;
        float xv = bf2f(qkv[row + d]);
        int   dp = (d < 64) ? d + 64 : d - 64;
        float xo = bf2f(qkv[row + dp]);
        float fi = (float)(2 * (d >> 1));
        float ang = (float)t * __expf(-fi * (9.210340371976184f / 128.0f));
        float c = __cosf(ang), s = __sinf(ang);
        float r = (d < 64) ? (xv * c - xo * s) : (xv * c + xo * s);
        Kr[((size_t)(b * NKV + kh) * SEQ + t) * HD + d] = f2bf(r);
    }
}

// V transpose: qkv cols 3072..4095 -> Vt[B,NKV,HD,T] (contiguous along T so
// the V A-operand loads in flash attention are contiguous b128s)
__global__ void k_vtrans(const unsigned short* __restrict__ qkv,
                         unsigned short* __restrict__ Vt) {
    const long n = (long)BATCH * SEQ * NKV * HD;
    long i = (long)blockIdx.x * blockDim.x + threadIdx.x;
    long st = (long)gridDim.x * blockDim.x;
    for (; i < n; i += st) {
        int d  = (int)(i & 127);
        int kh = (int)((i >> 7) & 7);
        int t  = (int)((i >> 10) & 2047);
        int b  = (int)(i >> 21);
        Vt[((size_t)(b * NKV + kh) * HD + d) * SEQ + t] =
            qkv[(size_t)(b * SEQ + t) * QKVN + 3072 + kh * HD + d];
    }
}

// ---------------------------------------------------------------------------
// bf16 GEMM:  D[M,N] = A[M,K] * W[N,K]^T   (A,W row-major bf16)
// 128x128 block tile, BK=32, 8 waves (4x2), each wave 2x4 WMMA C tiles.
// Double-buffered async global->LDS staging (ASYNCcnt), one barrier/K-step:
//   [wait asynccnt 0] [barrier] [issue async stage k+1] [8x WMMA on stage k]
// ---------------------------------------------------------------------------
template <int OUTF32>
__global__ __launch_bounds__(256)
void k_gemm(const unsigned short* __restrict__ A,
            const unsigned short* __restrict__ W,
            float* __restrict__ Df, unsigned short* __restrict__ Db,
            int M, int N, int K) {
    __shared__ __align__(16) unsigned short sA[2][128 * 40];
    __shared__ __align__(16) unsigned short sB[2][128 * 40];

    const int t    = threadIdx.x;
    const int lane = t & 31, wave = t >> 5;
    const int wm   = wave >> 1, wn = wave & 1;       // 4 x 2 wave grid
    const int l15  = lane & 15, h4 = lane >> 4;
    const long rowA0 = (long)blockIdx.y * 128;
    const long rowW0 = (long)blockIdx.x * 128;
    const int  chr  = t >> 2, chc = (t & 3) * 8;     // 16B chunk coords

    v8f acc[2][4] = {};

    // stage one 128x32 tile pair into buffer `buf` (4 async b128 per thread)
    auto stage = [&](int k0, int buf) {
        #pragma unroll
        for (int i = 0; i < 2; ++i) {
            int r = chr + i * 64;
            async_copy16(&A[(rowA0 + r) * K + k0 + chc], &sA[buf][r * 40 + chc]);
            async_copy16(&W[(rowW0 + r) * K + k0 + chc], &sB[buf][r * 40 + chc]);
        }
    };

    stage(0, 0);
    for (int k0 = 0; k0 < K; k0 += 32) {
        const int cur = (k0 >> 5) & 1;
        wait_async0();                               // stage k resident in LDS
        __syncthreads();                             // all threads done with buf cur^1
        if (k0 + 32 < K) stage(k0 + 32, cur ^ 1);    // overlaps the WMMAs below

        Frag32 fa[2], fb[4];
        #pragma unroll
        for (int mt = 0; mt < 2; ++mt) {             // A: lanes0-15 K0-7/16-23, lanes16-31 K8-15/24-31
            const unsigned short* p = &sA[cur][(wm * 32 + mt * 16 + l15) * 40 + 8 * h4];
            fa[mt].u[0] = *(const uint4*)p;
            fa[mt].u[1] = *(const uint4*)(p + 16);
        }
        #pragma unroll
        for (int nt = 0; nt < 4; ++nt) {             // B: lane=col, lanes0-15 K0-7/16-23, lanes16-31 K8-15/24-31
            const unsigned short* p = &sB[cur][(wn * 64 + nt * 16 + l15) * 40 + 8 * h4];
            fb[nt].u[0] = *(const uint4*)p;
            fb[nt].u[1] = *(const uint4*)(p + 16);
        }
        #pragma unroll
        for (int mt = 0; mt < 2; ++mt)
            #pragma unroll
            for (int nt = 0; nt < 4; ++nt)
                acc[mt][nt] = __builtin_amdgcn_wmma_f32_16x16x32_bf16(
                    false, fa[mt].v, false, fb[nt].v, (short)0, acc[mt][nt],
                    false, false);
    }

    #pragma unroll
    for (int mt = 0; mt < 2; ++mt)
        #pragma unroll
        for (int nt = 0; nt < 4; ++nt) {
            long mrow = rowA0 + wm * 32 + mt * 16 + 8 * h4;  // C: lane=col, rows M=8*h4+j
            long ncol = rowW0 + wn * 64 + nt * 16 + l15;
            #pragma unroll
            for (int j = 0; j < 8; ++j) {
                long off = (mrow + j) * (long)N + ncol;
                if (OUTF32) Df[off] = acc[mt][nt][j];
                else        Db[off] = f2bf(acc[mt][nt][j]);
            }
        }
}

// ---------------------------------------------------------------------------
// Flash attention (transposed-score formulation, all WMMA, GQA rep=2).
// Block: 8 waves x 16 queries = 128 queries. Per 32-key step per wave:
//   S^T(16k x 16q) = K_tile * Q^T   (8 WMMAs over hd=128, 2 key subtiles)
//   online softmax: in-lane reduce + one shfl_xor(16)
//   O^T(d x q)    += V^T_tile * P^T (8 WMMAs, one per 16-d tile)
// Causal mask compares only run on diagonal tiles (uniform branch).
// ---------------------------------------------------------------------------
__global__ __launch_bounds__(256)
void k_flash(const unsigned short* __restrict__ Qr,
             const unsigned short* __restrict__ Kr,
             const unsigned short* __restrict__ Vt,
             unsigned short* __restrict__ O) {
    const int qb = blockIdx.x, h = blockIdx.y, b = blockIdx.z;
    const int kvh  = h >> 1;
    const int lane = threadIdx.x & 31, wave = threadIdx.x >> 5;
    const int l15  = lane & 15, h4 = lane >> 4;
    const int qbase = qb * 128 + wave * 16;
    const int qg    = qbase + l15;                   // this lane's query column

    // Q^T B-fragments (loop-invariant): lane=query col, K(=d) 8h4..+7, 16+8h4..+7 per chunk
    const unsigned short* Qlane =
        Qr + ((size_t)(b * NH + h) * SEQ + qbase + l15) * HD + 8 * h4;
    Frag32 fq[4];
    #pragma unroll
    for (int c = 0; c < 4; ++c) {
        fq[c].u[0] = *(const uint4*)(Qlane + 32 * c);
        fq[c].u[1] = *(const uint4*)(Qlane + 32 * c + 16);
    }

    const unsigned short* Kb = Kr + (size_t)(b * NKV + kvh) * SEQ * HD;
    const unsigned short* Vb = Vt + (size_t)(b * NKV + kvh) * HD * SEQ;

    float m = -1e30f, lsum = 0.0f;
    v8f oacc[8] = {};

    for (int k0 = 0; k0 < qbase + 16; k0 += 32) {
        // --- S^T = K * Q^T, two 16-key subtiles ---
        v8f st[2] = {};
        #pragma unroll
        for (int s = 0; s < 2; ++s) {
            const unsigned short* kp =
                Kb + (size_t)(k0 + 16 * s + l15) * HD + 8 * h4;  // A: row=key
            #pragma unroll
            for (int c = 0; c < 4; ++c) {
                Frag32 fk;
                fk.u[0] = *(const uint4*)(kp + 32 * c);
                fk.u[1] = *(const uint4*)(kp + 32 * c + 16);
                st[s] = __builtin_amdgcn_wmma_f32_16x16x32_bf16(
                    false, fk.v, false, fq[c].v, (short)0, st[s], false, false);
            }
        }
        // --- causal mask: only diagonal tiles need compares (uniform branch) ---
        if (k0 + 32 > qbase) {
            const int krow = k0 + 8 * h4;            // C: lane holds keys 8*h4+j
            #pragma unroll
            for (int j = 0; j < 8; ++j) {
                if (krow + j      > qg) st[0][j] = -1e30f;
                if (krow + 16 + j > qg) st[1][j] = -1e30f;
            }
        }
        // --- online softmax ---
        float mt = -1e30f;
        #pragma unroll
        for (int j = 0; j < 8; ++j) mt = fmaxf(mt, fmaxf(st[0][j], st[1][j]));
        mt = fmaxf(mt, __shfl_xor(mt, 16, 32));
        float mnew  = fmaxf(m, mt);
        float alpha = __expf(m - mnew);
        float p0[8], p1[8], rs = 0.0f;
        #pragma unroll
        for (int j = 0; j < 8; ++j) {
            p0[j] = __expf(st[0][j] - mnew);
            p1[j] = __expf(st[1][j] - mnew);
            rs += p0[j] + p1[j];
        }
        rs += __shfl_xor(rs, 16, 32);
        lsum = lsum * alpha + rs;
        m = mnew;
        #pragma unroll
        for (int dt = 0; dt < 8; ++dt)
            #pragma unroll
            for (int j = 0; j < 8; ++j) oacc[dt][j] *= alpha;

        // --- repack P (C-layout) into P^T B-fragment: lane needs keys 16h4..16h4+15 ---
        float lo[8], hi[8];
        #pragma unroll
        for (int j = 0; j < 8; ++j) {
            float t0p = __shfl_xor(p0[j], 16, 32);
            float t1p = __shfl_xor(p1[j], 16, 32);
            lo[j] = h4 ? t1p : p0[j];                // keys 16h4+0..7
            hi[j] = h4 ? p1[j] : t0p;                // keys 16h4+8..15
        }
        Frag32 fp;
        fp.u[0] = make_uint4(packbf2(lo[0], lo[1]), packbf2(lo[2], lo[3]),
                             packbf2(lo[4], lo[5]), packbf2(lo[6], lo[7]));
        fp.u[1] = make_uint4(packbf2(hi[0], hi[1]), packbf2(hi[2], hi[3]),
                             packbf2(hi[4], hi[5]), packbf2(hi[6], hi[7]));

        // --- O^T += V^T * P^T, one WMMA per 16-row d tile ---
        #pragma unroll
        for (int dt = 0; dt < 8; ++dt) {
            const unsigned short* vp =
                Vb + (size_t)(dt * 16 + l15) * SEQ + k0 + 16 * h4;  // A: row=d
            Frag32 fv;
            fv.u[0] = *(const uint4*)vp;
            fv.u[1] = *(const uint4*)(vp + 8);
            oacc[dt] = __builtin_amdgcn_wmma_f32_16x16x32_bf16(
                false, fv.v, false, fp.v, (short)0, oacc[dt], false, false);
        }
    }

    // --- normalize and store O[B,T,NH,HD] (8 contiguous bf16 per tile) ---
    float inv = 1.0f / lsum;
    #pragma unroll
    for (int dt = 0; dt < 8; ++dt) {
        unsigned short* op =
            O + ((size_t)(b * SEQ + qbase + l15) * NH + h) * HD + dt * 16 + 8 * h4;
        uint4 w = make_uint4(
            packbf2(oacc[dt][0] * inv, oacc[dt][1] * inv),
            packbf2(oacc[dt][2] * inv, oacc[dt][3] * inv),
            packbf2(oacc[dt][4] * inv, oacc[dt][5] * inv),
            packbf2(oacc[dt][6] * inv, oacc[dt][7] * inv));
        *(uint4*)op = w;
    }
}

// ---------------------------------------------------------------------------
// launch
// ---------------------------------------------------------------------------
extern "C" void kernel_launch(void* const* d_in, const int* in_sizes, int n_in,
                              void* d_out, int out_size, void* d_ws, size_t ws_size,
                              hipStream_t stream) {
    (void)in_sizes; (void)n_in; (void)out_size; (void)ws_size;
    const float* x  = (const float*)d_in[0];
    const float* Wq = (const float*)d_in[1];
    const float* Wk = (const float*)d_in[2];
    const float* Wv = (const float*)d_in[3];
    const float* Wo = (const float*)d_in[4];

    char* ws = (char*)d_ws;
    const long MT = (long)BATCH * SEQ;               // 8192 rows
    // workspace layout (bytes); O aliases the dead QKV buffer
    unsigned short* xb   = (unsigned short*)(ws);                                   // 33.5 MB
    unsigned short* wqkv = (unsigned short*)(ws + 33554432);                        // 16.8 MB
    unsigned short* wo   = (unsigned short*)(ws + 50331648);                        //  8.4 MB
    unsigned short* qkv  = (unsigned short*)(ws + 58720256);                        // 67.1 MB
    unsigned short* qr   = (unsigned short*)(ws + 125829120);                       // 33.5 MB
    unsigned short* kr   = (unsigned short*)(ws + 159383552);                       // 16.8 MB
    unsigned short* vt   = (unsigned short*)(ws + 176160768);                       // 16.8 MB
    unsigned short* oatt = qkv;                                                     // alias

    const int CV = 256;
    // fp32 -> bf16: x and fused [Wq;Wk;Wv] (+Wo)
    k_cvt_bf16<<<8192, CV, 0, stream>>>(x,  xb,                 MT * CDIM);
    k_cvt_bf16<<<4096, CV, 0, stream>>>(Wq, wqkv,               (long)CDIM * CDIM);
    k_cvt_bf16<<<4096, CV, 0, stream>>>(Wk, wqkv + 4194304,     (long)1024 * CDIM);
    k_cvt_bf16<<<4096, CV, 0, stream>>>(Wv, wqkv + 6291456,     (long)1024 * CDIM);
    k_cvt_bf16<<<4096, CV, 0, stream>>>(Wo, wo,                 (long)CDIM * CDIM);

    // fused QKV projection: [8192,2048] x [4096,2048]^T -> bf16 [8192,4096]
    k_gemm<0><<<dim3(QKVN / 128, MT / 128), 256, 0, stream>>>(
        xb, wqkv, nullptr, qkv, (int)MT, QKVN, CDIM);

    // RoPE + layout permutes + V transpose
    k_rope_q<<<8192, CV, 0, stream>>>(qkv, qr);
    k_rope_k<<<8192, CV, 0, stream>>>(qkv, kr);
    k_vtrans<<<8192, CV, 0, stream>>>(qkv, vt);

    // flash attention -> O[B,T,C] bf16 (aliases qkv storage)
    k_flash<<<dim3(SEQ / 128, NH, BATCH), 256, 0, stream>>>(qr, kr, vt, oatt);

    // output projection: [8192,2048] x [2048,2048]^T -> fp32 d_out
    k_gemm<1><<<dim3(CDIM / 128, MT / 128), 256, 0, stream>>>(
        oatt, wo, (float*)d_out, nullptr, (int)MT, CDIM, CDIM);
}
